// PeerLayer_43748536877260
// MI455X (gfx1250) — compile-verified, matrix operand order checked by reading, not run on hardware
//
#include <hip/hip_runtime.h>
#include <hip/hip_bf16.h>
#include <math.h>

// ---------------------------------------------------------------------------
// PEER layer: q = x@Wq+bq ; product-key top-k routing ; gathered expert MLP
// B=2, S=1024 (2048 tokens), D_IN=D_OUT=1024, D_QUERY=256, H=8,
// N_SQRT=256, TOPK=10, DQ_HALF=128, N_EXPERTS=65536
// ---------------------------------------------------------------------------

typedef __attribute__((ext_vector_type(16))) __bf16 v16bf;
typedef __attribute__((ext_vector_type(8)))  __bf16 v8bf;
typedef __attribute__((ext_vector_type(8)))  float  v8f;
typedef int i32x4 __attribute__((vector_size(16)));

#define NTOK   2048
#define DIN    1024
#define NQ     2048      // D_QUERY * N_HEADS
#define NHEAD  8
#define DQ     256
#define DQH    128
#define NSQRT  256
#define TOPK   10
#define NSEL   (NHEAD * TOPK)     // 80 experts per token
#define NEG_INF (-1.0e30f)

#define BSTRIDE 40                // padded LDS row stride (bf16): 80B, 16B-aligned

#if defined(__has_builtin)
#  if __has_builtin(__builtin_amdgcn_global_load_async_to_lds_b128) && \
      __has_builtin(__builtin_amdgcn_s_wait_asynccnt)
#    define USE_ASYNC_LDS 1
#  endif
#endif
#ifndef USE_ASYNC_LDS
#  define USE_ASYNC_LDS 0
#endif

#if USE_ASYNC_LDS
// One 16B lane-slice: global -> LDS, tracked by ASYNCcnt.
__device__ __forceinline__ void async_copy_b128(const void* gsrc, void* ldst)
{
    __builtin_amdgcn_global_load_async_to_lds_b128(
        (__attribute__((address_space(1))) i32x4*)gsrc,
        (__attribute__((address_space(3))) i32x4*)ldst,
        0, 0);
}
#endif

// ============================================================
// Kernel 1: q = x @ Wq + bq   (M=2048, N=2048, K=1024) via WMMA
// block = 256 threads (8 waves); tile 16(M) x 256(N); K step 32
// each wave owns two 16x16 accumulators (A fragment reused)
// ============================================================
__global__ __launch_bounds__(256)
void qproj_wmma(const float* __restrict__ x, const float* __restrict__ Wq,
                const float* __restrict__ bq, float* __restrict__ q)
{
    // A tile row-major (K contiguous); B tile transposed (col-major, K contiguous)
    __shared__ __attribute__((aligned(16))) __bf16 As[16][BSTRIDE];
    __shared__ __attribute__((aligned(16))) __bf16 Bst[256][BSTRIDE];

    const int tid  = threadIdx.x;
    const int wave = tid >> 5;
    const int lane = tid & 31;
    const int tm   = blockIdx.x * 16;    // token tile base
    const int tn   = blockIdx.y * 256;   // output-dim tile base

    v8f acc0 = {};
    v8f acc1 = {};

    for (int k0 = 0; k0 < DIN; k0 += 32) {
        // ---- stage A tile: 16x32 fp32 -> bf16 (2 consecutive elems / thread) ----
        {
            const int i0 = tid * 2;
            const int r  = i0 >> 5;
            const int c  = i0 & 31;
            const float2 av =
                *reinterpret_cast<const float2*>(&x[(size_t)(tm + r) * DIN + k0 + c]);
            As[r][c]     = (__bf16)av.x;
            As[r][c + 1] = (__bf16)av.y;
        }
        // ---- stage B tile transposed: 32x256 fp32 -> Bst[col][k] bf16 ----
        // thread handles 4 cols x one k-pair per pass; packs (k,k+1) into one b32 store
        {
            unsigned* bw = reinterpret_cast<unsigned*>(&Bst[0][0]);
            #pragma unroll
            for (int j = 0; j < 4; ++j) {
                const int u  = tid + j * 256;      // 0..1023
                const int c  = (u << 2) & 255;     // col base
                const int kp = u >> 6;             // k-pair index 0..15
                const float4 b0 = *reinterpret_cast<const float4*>(
                    &Wq[(size_t)(k0 + 2 * kp) * NQ + tn + c]);
                const float4 b1 = *reinterpret_cast<const float4*>(
                    &Wq[(size_t)(k0 + 2 * kp + 1) * NQ + tn + c]);
                const float b0a[4] = {b0.x, b0.y, b0.z, b0.w};
                const float b1a[4] = {b1.x, b1.y, b1.z, b1.w};
                #pragma unroll
                for (int i = 0; i < 4; ++i) {
                    const unsigned lo =
                        (unsigned)__builtin_bit_cast(unsigned short, (__bf16)b0a[i]);
                    const unsigned hi =
                        (unsigned)__builtin_bit_cast(unsigned short, (__bf16)b1a[i]);
                    bw[(c + i) * (BSTRIDE / 2) + kp] = lo | (hi << 16);
                }
            }
        }
        __syncthreads();

        // ---- fragments: per ISA 16-bit layout, each fragment = two 16B LDS loads ----
        // lanes 0-15: K in {0..7, 16..23}; lanes 16-31: K in {8..15, 24..31}
        const int row = lane & 15;
        const int kb  = (lane >> 4) << 3;   // 0 or 8
        const v8bf alo = *reinterpret_cast<const v8bf*>(&As[row][kb]);
        const v8bf ahi = *reinterpret_cast<const v8bf*>(&As[row][16 + kb]);
        const v16bf afrag = __builtin_shufflevector(
            alo, ahi, 0, 1, 2, 3, 4, 5, 6, 7, 8, 9, 10, 11, 12, 13, 14, 15);

        {
            const int col0 = (wave << 5) + (lane & 15);
            const v8bf blo = *reinterpret_cast<const v8bf*>(&Bst[col0][kb]);
            const v8bf bhi = *reinterpret_cast<const v8bf*>(&Bst[col0][16 + kb]);
            const v16bf bfrag = __builtin_shufflevector(
                blo, bhi, 0, 1, 2, 3, 4, 5, 6, 7, 8, 9, 10, 11, 12, 13, 14, 15);
            acc0 = __builtin_amdgcn_wmma_f32_16x16x32_bf16(
                       false, afrag, false, bfrag, (short)0, acc0, false, false);
        }
        {
            const int col1 = (wave << 5) + 16 + (lane & 15);
            const v8bf blo = *reinterpret_cast<const v8bf*>(&Bst[col1][kb]);
            const v8bf bhi = *reinterpret_cast<const v8bf*>(&Bst[col1][16 + kb]);
            const v16bf bfrag = __builtin_shufflevector(
                blo, bhi, 0, 1, 2, 3, 4, 5, 6, 7, 8, 9, 10, 11, 12, 13, 14, 15);
            acc1 = __builtin_amdgcn_wmma_f32_16x16x32_bf16(
                       false, afrag, false, bfrag, (short)0, acc1, false, false);
        }
        __syncthreads();
    }

    // ---- store two 16x16 fp32 tiles (+bias) ----
    // C layout: VGPR r, lanes<16 -> M=r, lanes>=16 -> M=r+8; N = lane&15
    {
        const int n  = lane & 15;
        const int mh = (lane >> 4) << 3;
        {
            const int col = tn + (wave << 5) + n;
            const float bias = bq[col];
            #pragma unroll
            for (int r = 0; r < 8; ++r)
                q[(size_t)(tm + mh + r) * NQ + col] = acc0[r] + bias;
        }
        {
            const int col = tn + (wave << 5) + 16 + n;
            const float bias = bq[col];
            #pragma unroll
            for (int r = 0; r < 8; ++r)
                q[(size_t)(tm + mh + r) * NQ + col] = acc1[r] + bias;
        }
    }
}

// ============================================================
// Kernel 2: product-key scoring + two-level top-k + softmax
// one 128-thread block per (token, head) pair
// ============================================================
__device__ __forceinline__
void argmax10_256(float* s, float* vout, int* iout,
                  float* rv, int* ri, int tid)
{
    for (int it = 0; it < TOPK; ++it) {
        float v = s[tid]; int idx = tid;
        const float v2 = s[tid + 128];
        if (v2 > v) { v = v2; idx = tid + 128; }
        rv[tid] = v; ri[tid] = idx;
        __syncthreads();
        for (int off = 64; off > 0; off >>= 1) {
            if (tid < off && rv[tid + off] > rv[tid]) {
                rv[tid] = rv[tid + off]; ri[tid] = ri[tid + off];
            }
            __syncthreads();
        }
        if (tid == 0) { vout[it] = rv[0]; iout[it] = ri[0]; s[ri[0]] = NEG_INF; }
        __syncthreads();
    }
}

__global__ __launch_bounds__(128)
void score_topk(const float* __restrict__ q, const float* __restrict__ K1,
                const float* __restrict__ K2, int* __restrict__ sel_idx,
                float* __restrict__ sel_w)
{
    __shared__ float qs[256];
    __shared__ float s1[256];
    __shared__ float s2[256];
    __shared__ float v1[TOPK]; __shared__ int i1[TOPK];
    __shared__ float v2[TOPK]; __shared__ int i2[TOPK];
    __shared__ float rv[128];  __shared__ int ri[128];
    __shared__ float cv[128];  __shared__ int ci[128];
    __shared__ float ov[TOPK]; __shared__ int oi[TOPK];

    const int pair  = blockIdx.x;           // token*8 + head
    const int token = pair >> 3;
    const int head  = pair & 7;
    const int tid   = threadIdx.x;
    const float* qp = q + (size_t)token * NQ + head * DQ;

    qs[tid]       = qp[tid];
    qs[tid + 128] = qp[tid + 128];
    __syncthreads();

    // scores: s1[n] = q1 . K1[n],  s2[n] = q2 . K2[n]   (n = tid, tid+128)
    #pragma unroll
    for (int rep = 0; rep < 2; ++rep) {
        const int n = tid + rep * 128;
        const float* k1r = K1 + (size_t)n * DQH;
        const float* k2r = K2 + (size_t)n * DQH;
        float a1 = 0.f, a2 = 0.f;
        for (int d = 0; d < DQH; d += 4) {
            const float4 ka = *reinterpret_cast<const float4*>(k1r + d);
            const float4 kb = *reinterpret_cast<const float4*>(k2r + d);
            a1 += qs[d] * ka.x + qs[d + 1] * ka.y + qs[d + 2] * ka.z + qs[d + 3] * ka.w;
            a2 += qs[DQH + d] * kb.x + qs[DQH + d + 1] * kb.y +
                  qs[DQH + d + 2] * kb.z + qs[DQH + d + 3] * kb.w;
        }
        s1[n] = a1; s2[n] = a2;
    }
    __syncthreads();

    argmax10_256(s1, v1, i1, rv, ri, tid);
    argmax10_256(s2, v2, i2, rv, ri, tid);

    // 100 combos (padded to 128), then top-10 of combos
    if (tid < 100) {
        cv[tid] = v1[tid / 10] + v2[tid % 10];
        ci[tid] = i1[tid / 10] * NSQRT + i2[tid % 10];
    } else {
        cv[tid] = NEG_INF; ci[tid] = 0;
    }
    __syncthreads();

    for (int it = 0; it < TOPK; ++it) {
        rv[tid] = cv[tid]; ri[tid] = tid;
        __syncthreads();
        for (int off = 64; off > 0; off >>= 1) {
            if (tid < off && rv[tid + off] > rv[tid]) {
                rv[tid] = rv[tid + off]; ri[tid] = ri[tid + off];
            }
            __syncthreads();
        }
        if (tid == 0) { ov[it] = rv[0]; oi[it] = ci[ri[0]]; cv[ri[0]] = NEG_INF; }
        __syncthreads();
    }

    // softmax over the 10 selected (ov[0] is the max), write results
    if (tid == 0) {
        float sum = 0.f;
        float e[TOPK];
        for (int i = 0; i < TOPK; ++i) { e[i] = __expf(ov[i] - ov[0]); sum += e[i]; }
        const float inv = 1.0f / sum;
        const size_t base = (size_t)pair * TOPK;
        for (int i = 0; i < TOPK; ++i) {
            sel_idx[base + i] = oi[i];
            sel_w[base + i]   = e[i] * inv;
        }
    }
}

// ============================================================
// Kernel 3: gathered expert MLP (HBM-bound, ~1.3 GB of row gathers):
//   out = sum_{head,k} gelu(x . W_down[idx]) * w * W_up[idx]
// one 256-thread block per token. Double-buffered expert rows staged
// through LDS with GLOBAL_LOAD_ASYNC_TO_LDS_B128 (each lane copies its
// own 16B slice -> only the per-wave s_wait_asynccnt is needed).
// ============================================================
__global__ __launch_bounds__(256)
void expert_apply(const float* __restrict__ x, const float* __restrict__ W_up,
                  const float* __restrict__ W_down, const int* __restrict__ sel_idx,
                  const float* __restrict__ sel_w, float* __restrict__ out)
{
    __shared__ float wavesum[8];
    __shared__ float gbc;
#if USE_ASYNC_LDS
    __shared__ __attribute__((aligned(16))) float rowD[2][DIN];
    __shared__ __attribute__((aligned(16))) float rowU[2][DIN];
#endif

    const int token = blockIdx.x;
    const int tid   = threadIdx.x;
    const int lane  = tid & 31;
    const int wv    = tid >> 5;

    const float4 xr = *reinterpret_cast<const float4*>(x + (size_t)token * DIN + tid * 4);
    float4 acc = {0.f, 0.f, 0.f, 0.f};

    const int base = token * NSEL;   // 80 selections per token

#if USE_ASYNC_LDS
    {
        const int i0 = sel_idx[base];
        async_copy_b128(W_down + (size_t)i0 * DIN + tid * 4, &rowD[0][tid * 4]);
        async_copy_b128(W_up   + (size_t)i0 * DIN + tid * 4, &rowU[0][tid * 4]);
    }
#endif

    for (int e = 0; e < NSEL; ++e) {
        const float w = sel_w[base + e];
        float4 d, u;

#if USE_ASYNC_LDS
        const int buf = e & 1;
        if (e + 1 < NSEL) {
            const int ni = sel_idx[base + e + 1];
            async_copy_b128(W_down + (size_t)ni * DIN + tid * 4, &rowD[buf ^ 1][tid * 4]);
            async_copy_b128(W_up   + (size_t)ni * DIN + tid * 4, &rowU[buf ^ 1][tid * 4]);
            __builtin_amdgcn_s_wait_asynccnt(2);   // current buf's pair complete
        } else {
            __builtin_amdgcn_s_wait_asynccnt(0);
        }
        d = *reinterpret_cast<const float4*>(&rowD[buf][tid * 4]);
        u = *reinterpret_cast<const float4*>(&rowU[buf][tid * 4]);
#else
        const int idx = sel_idx[base + e];
        if (e + 1 < NSEL) {
            const int nidx = sel_idx[base + e + 1];
            __builtin_prefetch(W_down + (size_t)nidx * DIN + tid * 4, 0, 0);
            __builtin_prefetch(W_up   + (size_t)nidx * DIN + tid * 4, 0, 0);
        }
        d = *reinterpret_cast<const float4*>(W_down + (size_t)idx * DIN + tid * 4);
        u = *reinterpret_cast<const float4*>(W_up   + (size_t)idx * DIN + tid * 4);
#endif

        // h = x . W_down[idx]
        float p = xr.x * d.x + xr.y * d.y + xr.z * d.z + xr.w * d.w;
        #pragma unroll
        for (int off = 16; off > 0; off >>= 1) p += __shfl_down(p, off, 32);
        if (lane == 0) wavesum[wv] = p;
        __syncthreads();
        if (tid == 0) {
            float h = 0.f;
            #pragma unroll
            for (int i = 0; i < 8; ++i) h += wavesum[i];
            const float g = 0.5f * h * (1.0f + erff(h * 0.70710678118654752f)); // exact gelu
            gbc = g * w;
        }
        __syncthreads();
        const float g = gbc;

        acc.x += g * u.x; acc.y += g * u.y; acc.z += g * u.z; acc.w += g * u.w;
        __syncthreads();   // protect wavesum/gbc before next iteration
    }

    *reinterpret_cast<float4*>(out + (size_t)token * DIN + tid * 4) = acc;
}

// ============================================================
// launcher
// ============================================================
extern "C" void kernel_launch(void* const* d_in, const int* in_sizes, int n_in,
                              void* d_out, int out_size, void* d_ws, size_t ws_size,
                              hipStream_t stream)
{
    (void)in_sizes; (void)n_in; (void)out_size; (void)ws_size;

    const float* x      = (const float*)d_in[0];
    const float* Wq     = (const float*)d_in[1];
    const float* bq     = (const float*)d_in[2];
    const float* K1     = (const float*)d_in[3];
    const float* K2     = (const float*)d_in[4];
    const float* W_up   = (const float*)d_in[5];
    const float* W_down = (const float*)d_in[6];
    float* out = (float*)d_out;

    // workspace layout: q [2048*2048 f32] | sel_idx [16384*10 i32] | sel_w [16384*10 f32]
    char* ws = (char*)d_ws;
    float* q       = (float*)ws;
    int*   sel_idx = (int*)(ws + (size_t)NTOK * NQ * sizeof(float));
    float* sel_w   = (float*)(ws + (size_t)NTOK * NQ * sizeof(float)
                                 + (size_t)NTOK * NHEAD * TOPK * sizeof(int));

    // 1) q = x @ Wq + bq   (WMMA bf16 -> f32)
    dim3 g1(NTOK / 16, NQ / 256);
    qproj_wmma<<<g1, 256, 0, stream>>>(x, Wq, bq, q);

    // 2) product-key scoring + top-k routing
    score_topk<<<NTOK * NHEAD, 128, 0, stream>>>(q, K1, K2, sel_idx, sel_w);

    // 3) gathered expert down/up projection (HBM-bound, async LDS pipeline)
    expert_apply<<<NTOK, 256, 0, stream>>>(x, W_up, W_down, sel_idx, sel_w, out);
}
